// _ConformerLiteBlock_14542759264534
// MI455X (gfx1250) — matrix operand order repaired and checked by reading, not compiled, not run on hardware
//
#include <hip/hip_runtime.h>
#include <math.h>

// ---------------------------------------------------------------------------
// Problem constants (match reference)
// ---------------------------------------------------------------------------
static const int kB  = 8;
static const int kT  = 1024;
static const int kH  = 1024;
static const int kNH = 16;
static const int kM  = kB * kT;  // 8192 token rows

typedef __attribute__((ext_vector_type(16))) _Float16 v16h;
typedef __attribute__((ext_vector_type(4)))  _Float16 v4h;
typedef __attribute__((ext_vector_type(8)))  float    v8f;

__device__ __forceinline__ float gelu_exact(float x) {
    return 0.5f * x * (1.0f + erff(x * 0.70710678118654752f));
}

__device__ __forceinline__ v4h cvt4(float4 f) {
    v4h h;
    h[0] = (_Float16)f.x; h[1] = (_Float16)f.y;
    h[2] = (_Float16)f.z; h[3] = (_Float16)f.w;
    return h;
}

// ---------------------------------------------------------------------------
// WMMA fragment builders (wave32), per CDNA5 ISA layouts.
//  A (16x32 f16): lane -> m = L&15, hi = L>>4;
//    a[0..7] = A[m][hi*8 .. +8), a[8..15] = A[m][16+hi*8 .. +8)
//  B (32x16 f16): lane -> n = L&15, hi = L>>4; b[0..7] = B[hi*8 .. +8)[n], b[8..15] = B[16+hi*8 .. +8)[n]
//  (B staged transposed in LDS so each b-frag half is a contiguous 16B run.)
// ---------------------------------------------------------------------------
__device__ __forceinline__ v16h frag_a_row(const _Float16* row, int hi) {
    v16h a;
#pragma unroll
    for (int i = 0; i < 8; ++i) a[i] = row[hi * 8 + i];
#pragma unroll
    for (int i = 0; i < 8; ++i) a[8 + i] = row[16 + hi * 8 + i];
    return a;
}
__device__ __forceinline__ v16h frag_b_row(const _Float16* row, int hi) {
    v16h b;
#pragma unroll
    for (int j = 0; j < 16; ++j) b[j] = row[hi * 8 + (j & 7) + (j >> 3) * 16];
    return b;
}

// ---------------------------------------------------------------------------
// LayerNorm over last dim (H=1024).  One 256-thread block per row; each
// thread owns one float4.
// ---------------------------------------------------------------------------
__global__ __launch_bounds__(256) void ln_kernel(const float* __restrict__ x,
                                                 const float* __restrict__ g,
                                                 const float* __restrict__ b,
                                                 float* __restrict__ out) {
    __shared__ float red[256];
    const int tid = threadIdx.x;
    const size_t row = blockIdx.x;

    const float4 v = *(const float4*)(x + row * (size_t)kH + tid * 4);
    red[tid] = v.x + v.y + v.z + v.w;
    __syncthreads();
    for (int st = 128; st > 0; st >>= 1) {
        if (tid < st) red[tid] += red[tid + st];
        __syncthreads();
    }
    const float mean = red[0] * (1.0f / kH);
    __syncthreads();

    const float dx = v.x - mean, dy = v.y - mean, dz = v.z - mean, dw = v.w - mean;
    red[tid] = dx * dx + dy * dy + dz * dz + dw * dw;
    __syncthreads();
    for (int st = 128; st > 0; st >>= 1) {
        if (tid < st) red[tid] += red[tid + st];
        __syncthreads();
    }
    const float rstd = rsqrtf(red[0] * (1.0f / kH) + 1e-5f);

    const float4 gv = *(const float4*)(g + tid * 4);
    const float4 bv = *(const float4*)(b + tid * 4);
    float4 o;
    o.x = dx * rstd * gv.x + bv.x;
    o.y = dy * rstd * gv.y + bv.y;
    o.z = dz * rstd * gv.z + bv.z;
    o.w = dw * rstd * gv.w + bv.w;
    *(float4*)(out + row * (size_t)kH + tid * 4) = o;
}

// ---------------------------------------------------------------------------
// WMMA GEMM: out[M,N] = epilogue(A[M,K] @ B[K,N] + bias[N])
//   op: 0 = none, 1 = exact GELU, 2 = res + y, 3 = res + 0.5*y
//   btrans: B stored as [N,K] (pointwise conv weight)
// Block = 256 threads (8 waves); tile = 256(M) x 64(N); BK = 32.
// Wave w owns rows [w*32, w*32+32): 2 A-frags x 4 B-frags = 8 wmma / BK step.
// Global loads for step k+1 are issued under step k's wmma (register
// double-buffer software pipeline).
// ---------------------------------------------------------------------------
#define GBM 256
#define GBN 64
#define GBK 32

__global__ __launch_bounds__(256) void gemm_wmma(const float* __restrict__ A,
                                                 const float* __restrict__ Bw,
                                                 const float* __restrict__ bias,
                                                 const float* __restrict__ res,
                                                 float* __restrict__ out,
                                                 int M, int N, int K,
                                                 int op, int btrans) {
    __shared__ _Float16 As[GBM][GBK];      // 16 KB, row-major [m][k]
    __shared__ _Float16 Bs[GBN][GBK];      //  4 KB, transposed [n][k]

    const int tid  = threadIdx.x;
    const int lane = tid & 31;
    const int wave = tid >> 5;
    const int nl   = lane & 15;
    const int hi   = lane >> 4;
    const int m0   = blockIdx.y * GBM;
    const int n0   = blockIdx.x * GBN;

    float4 areg[8];   // 256x32 floats / 256 thr = 8 float4 each
    float4 breg[2];   //  64x32 floats / 256 thr = 2 float4 each

    auto stage = [&](int k0) {
#pragma unroll
        for (int i = 0; i < 8; ++i) {
            const int idx4 = tid + i * 256;
            const int r = idx4 >> 3, c4 = idx4 & 7;         // 8 float4 per row
            areg[i] = *(const float4*)(A + (size_t)(m0 + r) * K + k0 + c4 * 4);
        }
        if (btrans) {
#pragma unroll
            for (int i = 0; i < 2; ++i) {
                const int idx4 = tid + i * 256;
                const int nc = idx4 >> 3, c4 = idx4 & 7;    // coalesced over k
                breg[i] = *(const float4*)(Bw + (size_t)(n0 + nc) * K + k0 + c4 * 4);
            }
        } else {
#pragma unroll
            for (int i = 0; i < 2; ++i) {
                const int idx4 = tid + i * 256;
                const int kr = idx4 >> 4, n4 = idx4 & 15;   // coalesced over n
                breg[i] = *(const float4*)(Bw + (size_t)(k0 + kr) * N + n0 + n4 * 4);
            }
        }
    };
    auto commit = [&]() {
#pragma unroll
        for (int i = 0; i < 8; ++i) {
            const int idx4 = tid + i * 256;
            const int r = idx4 >> 3, c4 = idx4 & 7;
            *(v4h*)&As[r][c4 * 4] = cvt4(areg[i]);
        }
        if (btrans) {
#pragma unroll
            for (int i = 0; i < 2; ++i) {
                const int idx4 = tid + i * 256;
                const int nc = idx4 >> 3, c4 = idx4 & 7;
                *(v4h*)&Bs[nc][c4 * 4] = cvt4(breg[i]);     // direct: already [n][k]
            }
        } else {
#pragma unroll
            for (int i = 0; i < 2; ++i) {
                const int idx4 = tid + i * 256;
                const int kr = idx4 >> 4, n4 = idx4 & 15;
                Bs[n4 * 4 + 0][kr] = (_Float16)breg[i].x;   // transpose scatter
                Bs[n4 * 4 + 1][kr] = (_Float16)breg[i].y;
                Bs[n4 * 4 + 2][kr] = (_Float16)breg[i].z;
                Bs[n4 * 4 + 3][kr] = (_Float16)breg[i].w;
            }
        }
    };

    v8f acc[2][4] = {};
    stage(0);
    for (int k0 = 0; k0 < K; k0 += GBK) {
        commit();                         // waits on this step's loads only
        __syncthreads();
        if (k0 + GBK < K) stage(k0 + GBK);  // issue next-step loads now
        const v16h af0 = frag_a_row(&As[wave * 32 + nl][0], hi);
        const v16h af1 = frag_a_row(&As[wave * 32 + 16 + nl][0], hi);
#pragma unroll
        for (int t = 0; t < 4; ++t) {
            const v16h bf = frag_b_row(&Bs[t * 16 + nl][0], hi);
            acc[0][t] = __builtin_amdgcn_wmma_f32_16x16x32_f16(
                false, af0, false, bf, (short)0, acc[0][t], false, false);
            acc[1][t] = __builtin_amdgcn_wmma_f32_16x16x32_f16(
                false, af1, false, bf, (short)0, acc[1][t], false, false);
        }
        __syncthreads();
    }

    // Epilogue.  C layout: col = t*16 + nl; row (vgpr r) = hi*8 + r.
#pragma unroll
    for (int s = 0; s < 2; ++s)
#pragma unroll
        for (int t = 0; t < 4; ++t) {
            const int col = n0 + t * 16 + nl;
            const float bv = bias[col];
#pragma unroll
            for (int r = 0; r < 8; ++r) {
                const int row = m0 + wave * 32 + s * 16 + hi * 8 + r;
                float v = acc[s][t][r] + bv;
                if (op == 1) v = gelu_exact(v);
                const size_t idx = (size_t)row * N + col;
                if (op == 2)      v = res[idx] + v;
                else if (op == 3) v = res[idx] + 0.5f * v;
                out[idx] = v;
            }
        }
}

// ---------------------------------------------------------------------------
// Flash attention (causal).  One 128-thread block (4 waves) per
// (b, head, 64-row q tile).  Wave w owns q rows [q0+w*16, q0+w*16+16).
// K/V tiles for step kt+1 are loaded into registers under step kt's compute.
// qkv token layout: [3H] = { q: h*64+d, k: H+h*64+d, v: 2H+h*64+d }.
// ---------------------------------------------------------------------------
__global__ __launch_bounds__(128) void attn_wmma(const float* __restrict__ qkv,
                                                 float* __restrict__ o) {
    __shared__ _Float16 Qs[64][64];        // [q][d], pre-scaled by 1/8
    __shared__ _Float16 Ks[64][64];        // [k][d]
    __shared__ _Float16 Vt[64][64];        // [d][k] (transposed V)
    __shared__ _Float16 Ps[4][16][64];     // per-wave P scratch [q][k]

    const int tid  = threadIdx.x;
    const int lane = tid & 31;
    const int wave = tid >> 5;
    const int nl   = lane & 15;
    const int hi   = lane >> 4;

    const int qt = blockIdx.x % (kT / 64);
    const int h  = (blockIdx.x / (kT / 64)) % kNH;
    const int b  = blockIdx.x / ((kT / 64) * kNH);
    const int q0 = qt * 64;
    const float NEG = -__builtin_inff();

    float4 kreg[8], vreg[8];
    auto stage_kv = [&](int kt) {
        const int k0 = kt * 64;
#pragma unroll
        for (int i = 0; i < 8; ++i) {
            const int idx4 = tid + i * 128;
            const int r = idx4 >> 4, d4 = idx4 & 15;   // 16 float4 per 64-row
            const size_t base = (size_t)(b * kT + k0 + r) * (3 * kH) + h * 64 + d4 * 4;
            kreg[i] = *(const float4*)(qkv + base + kH);
            vreg[i] = *(const float4*)(qkv + base + 2 * kH);
        }
    };
    auto commit_kv = [&]() {
#pragma unroll
        for (int i = 0; i < 8; ++i) {
            const int idx4 = tid + i * 128;
            const int r = idx4 >> 4, d4 = idx4 & 15;
            *(v4h*)&Ks[r][d4 * 4] = cvt4(kreg[i]);
            Vt[d4 * 4 + 0][r] = (_Float16)vreg[i].x;   // transpose scatter
            Vt[d4 * 4 + 1][r] = (_Float16)vreg[i].y;
            Vt[d4 * 4 + 2][r] = (_Float16)vreg[i].z;
            Vt[d4 * 4 + 3][r] = (_Float16)vreg[i].w;
        }
    };

    // Q tile (scaled by 1/sqrt(HD) = 0.125), vectorized.
#pragma unroll
    for (int i = 0; i < 8; ++i) {
        const int idx4 = tid + i * 128;
        const int r = idx4 >> 4, d4 = idx4 & 15;
        float4 q = *(const float4*)(qkv + (size_t)(b * kT + q0 + r) * (3 * kH) + h * 64 + d4 * 4);
        q.x *= 0.125f; q.y *= 0.125f; q.z *= 0.125f; q.w *= 0.125f;
        *(v4h*)&Qs[r][d4 * 4] = cvt4(q);
    }

    float rmax[8], rsum[8];
    v8f oacc[4] = {};
#pragma unroll
    for (int r = 0; r < 8; ++r) { rmax[r] = NEG; rsum[r] = 0.f; }

    stage_kv(0);
    for (int kt = 0; kt <= qt; ++kt) {
        const int k0 = kt * 64;
        __syncthreads();     // prior K/V reads complete (and Q store ordering)
        commit_kv();
        __syncthreads();     // tiles visible
        if (kt < qt) stage_kv(kt + 1);   // prefetch next K/V under compute

        // S = Q @ K^T (contract d in two 32-chunks)
        v8f s[4] = {};
#pragma unroll
        for (int dd = 0; dd < 64; dd += 32) {
            const v16h aq = frag_a_row(&Qs[wave * 16 + nl][dd], hi);
#pragma unroll
            for (int t = 0; t < 4; ++t) {
                const v16h bk = frag_b_row(&Ks[t * 16 + nl][dd], hi);
                s[t] = __builtin_amdgcn_wmma_f32_16x16x32_f16(
                    false, aq, false, bk, (short)0, s[t], false, false);
            }
        }

        // Causal mask: only the diagonal tile straddles the boundary.
        if (k0 == q0) {
#pragma unroll
            for (int t = 0; t < 4; ++t)
#pragma unroll
                for (int r = 0; r < 8; ++r) {
                    const int q = q0 + wave * 16 + hi * 8 + r;
                    const int k = k0 + t * 16 + nl;
                    if (k > q) s[t][r] = NEG;
                }
        }

        // Online softmax stats (row spans 4 tiles x 16 lanes of a half-wave).
        float fac[8];
#pragma unroll
        for (int r = 0; r < 8; ++r) {
            float tm = s[0][r];
#pragma unroll
            for (int t = 1; t < 4; ++t) tm = fmaxf(tm, s[t][r]);
            for (int off = 1; off < 16; off <<= 1)
                tm = fmaxf(tm, __shfl_xor(tm, off, 16));
            const float nm = fmaxf(rmax[r], tm);
            fac[r] = expf(rmax[r] - nm);    // 0 on first tile (exp(-inf))
            rmax[r] = nm;
        }

        // P = exp(S - m); stash to per-wave LDS (C-layout -> A-layout).
#pragma unroll
        for (int t = 0; t < 4; ++t)
#pragma unroll
            for (int r = 0; r < 8; ++r) {
                const float p = expf(s[t][r] - rmax[r]);
                s[t][r] = p;
                Ps[wave][hi * 8 + r][t * 16 + nl] = (_Float16)p;
            }

        // Row sums + rescale O.
#pragma unroll
        for (int r = 0; r < 8; ++r) {
            float ps = s[0][r] + s[1][r] + s[2][r] + s[3][r];
            for (int off = 1; off < 16; off <<= 1)
                ps += __shfl_xor(ps, off, 16);
            rsum[r] = rsum[r] * fac[r] + ps;
#pragma unroll
            for (int t = 0; t < 4; ++t) oacc[t][r] *= fac[r];
        }

        // O += P @ V (same-wave LDS ops are in-order: no barrier needed).
#pragma unroll
        for (int kk = 0; kk < 64; kk += 32) {
            const v16h ap = frag_a_row(&Ps[wave][nl][kk], hi);
#pragma unroll
            for (int t = 0; t < 4; ++t) {
                const v16h bv = frag_b_row(&Vt[t * 16 + nl][kk], hi);
                oacc[t] = __builtin_amdgcn_wmma_f32_16x16x32_f16(
                    false, ap, false, bv, (short)0, oacc[t], false, false);
            }
        }
    }

    // O / l -> o[b, q, h*64 + d]
#pragma unroll
    for (int t = 0; t < 4; ++t)
#pragma unroll
        for (int r = 0; r < 8; ++r) {
            const int q = q0 + wave * 16 + hi * 8 + r;
            const int d = t * 16 + nl;
            o[(size_t)(b * kT + q) * kH + h * 64 + d] = oacc[t][r] / rsum[r];
        }
}

// ---------------------------------------------------------------------------
// Depthwise causal conv (K=3) + bias + exact GELU, on (B,T,H) fp32.
// ---------------------------------------------------------------------------
__global__ __launch_bounds__(256) void dwconv_gelu(const float* __restrict__ cn,
                                                   const float* __restrict__ wk,
                                                   const float* __restrict__ bias,
                                                   float* __restrict__ out) {
    const size_t idx = (size_t)blockIdx.x * blockDim.x + threadIdx.x;
    const size_t total = (size_t)kB * kT * kH;
    if (idx >= total) return;
    const int ch = (int)(idx % kH);
    const int t  = (int)((idx / kH) % kT);

    float acc = bias[ch];
    const float* p = cn + idx;
#pragma unroll
    for (int j = 0; j < 3; ++j) {
        const int ts = t - 2 + j;
        if (ts >= 0) acc += p[(j - 2) * kH] * wk[ch * 3 + j];
    }
    out[idx] = gelu_exact(acc);
}

// ---------------------------------------------------------------------------
// Driver
// ---------------------------------------------------------------------------
extern "C" void kernel_launch(void* const* d_in, const int* in_sizes, int n_in,
                              void* d_out, int out_size, void* d_ws, size_t ws_size,
                              hipStream_t stream) {
    const float* x       = (const float*)d_in[0];
    const float* ln1_g   = (const float*)d_in[1];
    const float* ln1_b   = (const float*)d_in[2];
    const float* ffn1_w1 = (const float*)d_in[3];
    const float* ffn1_b1 = (const float*)d_in[4];
    const float* ffn1_w2 = (const float*)d_in[5];
    const float* ffn1_b2 = (const float*)d_in[6];
    const float* an_g    = (const float*)d_in[7];
    const float* an_b    = (const float*)d_in[8];
    const float* qkv_w   = (const float*)d_in[9];
    const float* qkv_b   = (const float*)d_in[10];
    const float* out_w   = (const float*)d_in[11];
    const float* out_b   = (const float*)d_in[12];
    const float* cn_g    = (const float*)d_in[13];
    const float* cn_b    = (const float*)d_in[14];
    const float* dw_k    = (const float*)d_in[15];
    const float* dw_b    = (const float*)d_in[16];
    const float* pw_w    = (const float*)d_in[17];
    const float* pw_b    = (const float*)d_in[18];
    const float* ln2_g   = (const float*)d_in[19];
    const float* ln2_b   = (const float*)d_in[20];
    const float* ffn2_w1 = (const float*)d_in[21];
    const float* ffn2_b1 = (const float*)d_in[22];
    const float* ffn2_w2 = (const float*)d_in[23];
    const float* ffn2_b2 = (const float*)d_in[24];
    float* out = (float*)d_out;

    // Workspace: A 32MB | B 96MB | C 32MB | D 32MB
    char* ws = (char*)d_ws;
    float* bufA = (float*)(ws);
    float* bufB = (float*)(ws + 33554432ull);
    float* bufC = (float*)(ws + 134217728ull);
    float* bufD = (float*)(ws + 167772160ull);

    const dim3 blk256(256);
    const dim3 gM(kM);                       // LN: one block per row
    const dim3 gGemm1(1024 / GBN, kM / GBM); // N=1024
    const dim3 gGemm2(2048 / GBN, kM / GBM); // N=2048
    const dim3 gGemm3(3072 / GBN, kM / GBM); // N=3072

    // ---- x = x + 0.5 * FFN1(LN(x)) ----
    ln_kernel<<<gM, blk256, 0, stream>>>(x, ln1_g, ln1_b, bufA);
    gemm_wmma<<<gGemm2, blk256, 0, stream>>>(bufA, ffn1_w1, ffn1_b1, nullptr,
                                             bufB, kM, 2048, 1024, /*op=*/1, 0);
    gemm_wmma<<<gGemm1, blk256, 0, stream>>>(bufB, ffn1_w2, ffn1_b2, x,
                                             bufC, kM, 1024, 2048, /*op=*/3, 0);

    // ---- attention ----
    ln_kernel<<<gM, blk256, 0, stream>>>(bufC, an_g, an_b, bufA);
    gemm_wmma<<<gGemm3, blk256, 0, stream>>>(bufA, qkv_w, qkv_b, nullptr,
                                             bufB, kM, 3072, 1024, /*op=*/0, 0);
    attn_wmma<<<kB * kNH * (kT / 64), 128, 0, stream>>>(bufB, bufA);
    gemm_wmma<<<gGemm1, blk256, 0, stream>>>(bufA, out_w, out_b, bufC,
                                             bufD, kM, 1024, 1024, /*op=*/2, 0);

    // ---- conv module ----
    ln_kernel<<<gM, blk256, 0, stream>>>(bufD, cn_g, cn_b, bufA);
    dwconv_gelu<<<(kB * kT * kH) / 256, blk256, 0, stream>>>(bufA, dw_k, dw_b, bufC);
    gemm_wmma<<<gGemm1, blk256, 0, stream>>>(bufC, pw_w, pw_b, bufD,
                                             bufA, kM, 1024, 1024, /*op=*/2, /*btrans=*/1);

    // ---- x = x + 0.5 * FFN2(LN(x)) ----
    ln_kernel<<<gM, blk256, 0, stream>>>(bufA, ln2_g, ln2_b, bufC);
    gemm_wmma<<<gGemm2, blk256, 0, stream>>>(bufC, ffn2_w1, ffn2_b1, nullptr,
                                             bufB, kM, 2048, 1024, /*op=*/1, 0);
    gemm_wmma<<<gGemm1, blk256, 0, stream>>>(bufB, ffn2_w2, ffn2_b2, bufA,
                                             out, kM, 1024, 2048, /*op=*/3, 0);
}
